// Attention_52690658787659
// MI455X (gfx1250) — matrix operand order repaired and checked by reading, not compiled, not run on hardware
//
#include <hip/hip_runtime.h>

typedef __attribute__((ext_vector_type(16))) __bf16 v16bf;
typedef __attribute__((ext_vector_type(8)))  __bf16 v8bf;
typedef __attribute__((ext_vector_type(8)))  float  v8f;

namespace {
constexpr int Bc = 8;
constexpr int Tc = 2048;
constexpr int Hc = 512;
constexpr int Ec = 512;
constexpr int BT = Bc * Tc; // 16384
}

union V16U { v16bf v; v8bf h[2]; };

// ---------------------------------------------------------------------------
// LDS-staged, async double-buffered block GEMM.
// Block = 256 threads (8 waves), block tile 128(M) x 128(N), wave tile 32x64.
// A panel 128x32 bf16 (8KB) + B panel 128x32 bf16 (8KB), double buffered.
// Panels filled with GLOBAL_LOAD_ASYNC_TO_LDS_B128 (ASYNCcnt), 4 instr/wave/stage.
// ---------------------------------------------------------------------------
__device__ __forceinline__ void lds_gemm_128x128(
    const __bf16* __restrict__ A, int lda, int m0,
    const __bf16* __restrict__ Bt, int ldb, int n0,
    int K, __bf16* As, __bf16* Bs, v8f c[2][4])
{
  const int tid  = threadIdx.x;
  const int lane = tid & 31;
  const int wid  = tid >> 5;
  const int hf   = lane >> 4;
  const int lm   = lane & 15;
  const int m0w  = (wid & 3) * 32;   // wave M offset in block tile
  const int n0w  = (wid >> 2) * 64;  // wave N offset in block tile
  const int NK   = K / 32;

  const uint32_t ldsA = (uint32_t)(uintptr_t)As;  // low 32 bits = LDS offset
  const uint32_t ldsB = (uint32_t)(uintptr_t)Bs;

  auto issue_stage = [&](int buf, int kb) {
    const int k0 = kb * 32;
#pragma unroll
    for (int c2 = 0; c2 < 2; ++c2) {
      const int q   = tid * 2 + c2;      // 16B chunk id, 0..511
      const int r   = q >> 2;            // panel row 0..127
      const int col = q & 3;             // 16B chunk within 64B row
      uint32_t la = ldsA + (uint32_t)(buf * 8192 + q * 16);
      uint64_t ga = (uint64_t)(uintptr_t)(A + (size_t)(m0 + r) * lda + k0 + col * 8);
      asm volatile("global_load_async_to_lds_b128 %0, %1, off"
                   :: "v"(la), "v"(ga) : "memory");
      uint32_t lb = ldsB + (uint32_t)(buf * 8192 + q * 16);
      uint64_t gb = (uint64_t)(uintptr_t)(Bt + (size_t)(n0 + r) * ldb + k0 + col * 8);
      asm volatile("global_load_async_to_lds_b128 %0, %1, off"
                   :: "v"(lb), "v"(gb) : "memory");
    }
  };

  issue_stage(0, 0);
  for (int kb = 0; kb < NK; ++kb) {
    if (kb + 1 < NK) {
      issue_stage((kb + 1) & 1, kb + 1);
      asm volatile("s_wait_asynccnt 0x4" ::: "memory"); // stage kb landed (per-wave)
    } else {
      asm volatile("s_wait_asynccnt 0x0" ::: "memory");
    }
    __syncthreads();  // all waves' copies visible; prev readers done with this buffer

    const __bf16* Ap = As + (kb & 1) * 4096;  // elements
    const __bf16* Bp = Bs + (kb & 1) * 4096;

    V16U b[4];
#pragma unroll
    for (int nj = 0; nj < 4; ++nj) {
      const __bf16* br = Bp + (size_t)(n0w + nj * 16 + lm) * 32 + 16 * hf;
      b[nj].h[0] = *(const v8bf*)(br);
      b[nj].h[1] = *(const v8bf*)(br + 8);
    }
#pragma unroll
    for (int mi = 0; mi < 2; ++mi) {
      const __bf16* ar = Ap + (size_t)(m0w + mi * 16 + lm) * 32;
      V16U a;
      a.h[0] = *(const v8bf*)(ar + 16 * hf);
      a.h[1] = *(const v8bf*)(ar + 16 * hf + 8);
#pragma unroll
      for (int nj = 0; nj < 4; ++nj)
        c[mi][nj] = __builtin_amdgcn_wmma_f32_16x16x32_bf16(
            false, a.v, false, b[nj].v, (short)0, c[mi][nj], false, false);
    }
    __syncthreads();  // compute done before next iteration overwrites buffer
  }
}

// ---------------- conversion kernels ----------------

__global__ __launch_bounds__(256) void cvt_tokens_kernel(
    const float* __restrict__ src, __bf16* __restrict__ dst)
{
  size_t i = (size_t)blockIdx.x * 256 + threadIdx.x;
  dst[i] = (__bf16)src[i];
}

__global__ void cvt_w_kernel(
    const float* __restrict__ Wq, const float* __restrict__ Wk, const float* __restrict__ Wv,
    __bf16* __restrict__ Wqt, __bf16* __restrict__ Wkt, __bf16* __restrict__ Wvt)
{
  int h = blockIdx.x * 16 + threadIdx.x;
  int e = blockIdx.y * 16 + threadIdx.y;
  const float* src = (blockIdx.z == 0) ? Wq : (blockIdx.z == 1) ? Wk : Wv;
  __bf16* dst      = (blockIdx.z == 0) ? Wqt : (blockIdx.z == 1) ? Wkt : Wvt;
  dst[(size_t)e * Hc + h] = (__bf16)src[(size_t)h * Ec + e];
}

// ---------------- QKV projection (LDS/async path) ----------------
// C[BT x 512] = Tb @ W; z selects (Wq->Qb, Wk->Kb, Wv->Vt transposed per batch)
__global__ __launch_bounds__(256) void proj_kernel(
    const __bf16* __restrict__ Tb,
    const __bf16* __restrict__ Wqt, const __bf16* __restrict__ Wkt, const __bf16* __restrict__ Wvt,
    __bf16* __restrict__ Qb, __bf16* __restrict__ Kb, __bf16* __restrict__ Vt)
{
  __shared__ __bf16 As[2 * 128 * 32];
  __shared__ __bf16 Bs[2 * 128 * 32];

  const int lane = threadIdx.x & 31;
  const int wid  = threadIdx.x >> 5;
  const int hf   = lane >> 4;
  const int lm   = lane & 15;
  const int m0 = blockIdx.y * 128;   // row in [0, BT)
  const int n0 = blockIdx.x * 128;   // col in [0, 512)
  const int z  = blockIdx.z;

  const __bf16* Btp = (z == 0) ? Wqt : (z == 1) ? Wkt : Wvt;

  v8f c[2][4] = {};
  lds_gemm_128x128(Tb, Hc, m0, Btp, Hc, n0, Hc, As, Bs, c);

  const int m0w = (wid & 3) * 32, n0w = (wid >> 2) * 64;
  if (z < 2) {
    __bf16* out = (z == 0) ? Qb : Kb;
#pragma unroll
    for (int mi = 0; mi < 2; ++mi)
#pragma unroll
      for (int nj = 0; nj < 4; ++nj)
#pragma unroll
        for (int r = 0; r < 8; ++r) {
          int m = m0 + m0w + mi * 16 + r + 8 * hf;
          int n = n0 + n0w + nj * 16 + lm;
          out[(size_t)m * Ec + n] = (__bf16)c[mi][nj][r];
        }
  } else {
    // V stored transposed: Vt[b][h][s]  (128-row tiles never straddle a batch)
#pragma unroll
    for (int mi = 0; mi < 2; ++mi)
#pragma unroll
      for (int nj = 0; nj < 4; ++nj)
#pragma unroll
        for (int r = 0; r < 8; ++r) {
          int m = m0 + m0w + mi * 16 + r + 8 * hf;
          int n = n0 + n0w + nj * 16 + lm;
          int b = m >> 11;
          int s = m & (Tc - 1);
          Vt[((size_t)b * Hc + n) * Tc + s] = (__bf16)c[mi][nj][r];
        }
  }
}

// ---------------- scores: S[b] = Q[b] @ K[b]^T (f32), LDS/async path ----------------
__global__ __launch_bounds__(256) void scores_kernel(
    const __bf16* __restrict__ Qb, const __bf16* __restrict__ Kb, float* __restrict__ S)
{
  __shared__ __bf16 As[2 * 128 * 32];
  __shared__ __bf16 Bs[2 * 128 * 32];

  const int lane = threadIdx.x & 31;
  const int wid  = threadIdx.x >> 5;
  const int hf   = lane >> 4;
  const int lm   = lane & 15;
  const int m0 = blockIdx.y * 128;
  const int n0 = blockIdx.x * 128;
  const int b  = blockIdx.z;

  const __bf16* A  = Qb + (size_t)b * Tc * Ec;
  const __bf16* Bt = Kb + (size_t)b * Tc * Ec;   // row n holds K[s=n][e], K-contiguous

  v8f c[2][4] = {};
  lds_gemm_128x128(A, Ec, m0, Bt, Ec, n0, Ec, As, Bs, c);

  float* out = S + (size_t)b * Tc * Tc;
  const int m0w = (wid & 3) * 32, n0w = (wid >> 2) * 64;
#pragma unroll
  for (int mi = 0; mi < 2; ++mi)
#pragma unroll
    for (int nj = 0; nj < 4; ++nj)
#pragma unroll
      for (int r = 0; r < 8; ++r) {
        int m = m0 + m0w + mi * 16 + r + 8 * hf;
        int n = n0 + n0w + nj * 16 + lm;
        out[(size_t)m * Tc + n] = c[mi][nj][r];
      }
}

// ---------------- column softmax stats (softmax over axis=1 == query axis t) ----------------
__global__ __launch_bounds__(256) void colstats_kernel(
    const float* __restrict__ S, float* __restrict__ mstat, float* __restrict__ rstat)
{
  int idx = blockIdx.x * 256 + threadIdx.x;   // over B*T columns
  int b = idx >> 11;
  int s = idx & (Tc - 1);
  const float* col = S + (size_t)b * Tc * Tc + s;
  float m = -3.402823466e38f;
  for (int t = 0; t < Tc; ++t) m = fmaxf(m, col[(size_t)t * Tc]);
  float sum = 0.f;
  for (int t = 0; t < Tc; ++t) sum += __expf(col[(size_t)t * Tc] - m);
  mstat[idx] = m;
  rstat[idx] = 1.f / sum;
}

__global__ __launch_bounds__(256) void expo_kernel(
    const float* __restrict__ S, const float* __restrict__ mstat, __bf16* __restrict__ Ab)
{
  size_t i = (size_t)blockIdx.x * 256 + threadIdx.x; // over B*T*T
  int s = (int)(i & (size_t)(Tc - 1));
  int b = (int)(i >> 22);
  Ab[i] = (__bf16)__expf(S[i] - mstat[(b << 11) | s]);
}

__global__ __launch_bounds__(256) void vscale_kernel(
    __bf16* __restrict__ Vt, const float* __restrict__ rstat)
{
  size_t i = (size_t)blockIdx.x * 256 + threadIdx.x; // over B*H*T
  int s = (int)(i & (size_t)(Tc - 1));
  int b = (int)(i >> 20);
  Vt[i] = (__bf16)((float)Vt[i] * rstat[(b << 11) | s]);
}

// ---------------- ctx[b] = Ab[b] @ V'[b], LDS/async path ----------------
__global__ __launch_bounds__(256) void ctx_kernel(
    const __bf16* __restrict__ Ab, const __bf16* __restrict__ Vt, float* __restrict__ out)
{
  __shared__ __bf16 As[2 * 128 * 32];
  __shared__ __bf16 Bs[2 * 128 * 32];

  const int lane = threadIdx.x & 31;
  const int wid  = threadIdx.x >> 5;
  const int hf   = lane >> 4;
  const int lm   = lane & 15;
  const int m0 = blockIdx.y * 128;  // t
  const int n0 = blockIdx.x * 128;  // h
  const int b  = blockIdx.z;

  const __bf16* A  = Ab + (size_t)b * Tc * Tc;   // [t][s], K=s contiguous
  const __bf16* Bt = Vt + (size_t)b * Hc * Tc;   // row n=h holds V'[s][h] over s

  v8f c[2][4] = {};
  lds_gemm_128x128(A, Tc, m0, Bt, Tc, n0, Tc, As, Bs, c);

  float* o = out + (size_t)b * Tc * Hc;
  const int m0w = (wid & 3) * 32, n0w = (wid >> 2) * 64;
#pragma unroll
  for (int mi = 0; mi < 2; ++mi)
#pragma unroll
    for (int nj = 0; nj < 4; ++nj)
#pragma unroll
      for (int r = 0; r < 8; ++r) {
        int m = m0 + m0w + mi * 16 + r + 8 * hf;
        int n = n0 + n0w + nj * 16 + lm;
        o[(size_t)m * Hc + n] = c[mi][nj][r];
      }
}

// ---------------- launch ----------------

extern "C" void kernel_launch(void* const* d_in, const int* in_sizes, int n_in,
                              void* d_out, int out_size, void* d_ws, size_t ws_size,
                              hipStream_t stream) {
  const float* tokens = (const float*)d_in[0];
  const float* Wq     = (const float*)d_in[1];
  const float* Wk     = (const float*)d_in[2];
  const float* Wv     = (const float*)d_in[3];
  float* out = (float*)d_out;

  char* w = (char*)d_ws;
  size_t off = 0;
  auto alloc = [&](size_t bytes) { size_t o = off; off += (bytes + 255) & ~(size_t)255; return o; };

  __bf16* Tb   = (__bf16*)(w + alloc((size_t)BT * Hc * 2));
  __bf16* Wqt  = (__bf16*)(w + alloc((size_t)Ec * Hc * 2));
  __bf16* Wkt  = (__bf16*)(w + alloc((size_t)Ec * Hc * 2));
  __bf16* Wvt  = (__bf16*)(w + alloc((size_t)Hc * Hc * 2));
  __bf16* Qb   = (__bf16*)(w + alloc((size_t)BT * Ec * 2));
  __bf16* Kb   = (__bf16*)(w + alloc((size_t)BT * Ec * 2));
  __bf16* Vt   = (__bf16*)(w + alloc((size_t)Bc * Hc * Tc * 2));
  float*  S    = (float*)(w + alloc((size_t)Bc * Tc * Tc * 4));
  __bf16* Ab   = (__bf16*)(w + alloc((size_t)Bc * Tc * Tc * 2));
  float* mstat = (float*)(w + alloc((size_t)BT * 4));
  float* rstat = (float*)(w + alloc((size_t)BT * 4));

  cvt_tokens_kernel<<<(size_t)BT * Hc / 256, 256, 0, stream>>>(tokens, Tb);
  cvt_w_kernel<<<dim3(Hc / 16, Ec / 16, 3), dim3(16, 16, 1), 0, stream>>>(
      Wq, Wk, Wv, Wqt, Wkt, Wvt);
  proj_kernel<<<dim3(Ec / 128, BT / 128, 3), 256, 0, stream>>>(Tb, Wqt, Wkt, Wvt, Qb, Kb, Vt);
  scores_kernel<<<dim3(Tc / 128, Tc / 128, Bc), 256, 0, stream>>>(Qb, Kb, S);
  colstats_kernel<<<BT / 256, 256, 0, stream>>>(S, mstat, rstat);
  expo_kernel<<<(size_t)Bc * Tc * Tc / 256, 256, 0, stream>>>(S, mstat, Ab);
  vscale_kernel<<<(size_t)Bc * Hc * Tc / 256, 256, 0, stream>>>(Vt, rstat);
  ctx_kernel<<<dim3(Hc / 128, Tc / 128, Bc), 256, 0, stream>>>(Ab, Vt, out);
}
